// ContrastiveFeatureSeparator_1511828488310
// MI455X (gfx1250) — matrix operand order repaired and checked by reading, not compiled, not run on hardware
//
#include <hip/hip_runtime.h>
#include <hip/hip_bf16.h>
#include <math.h>

typedef __bf16 bf16;
typedef __attribute__((ext_vector_type(16))) bf16  v16bf;
typedef __attribute__((ext_vector_type(8)))  float v8f;

// ---- geometry (fixed by the reference) ----
#define NIMG 10        // B*M
#define CCH  128       // channels
#define HWPX 65536     // 256*256
#define WPIX 256
#define NGRP 8
#define CPG  16        // channels per group
#define LSTR 136       // LDS row stride in bf16 (272B = 17*16B -> 16B aligned rows)

union FragU { uint4 q[2]; v16bf v; };

// A fragment (16x32 bf16): lane row = lane&15; elems 0..15 = k..k+15 contiguous
__device__ __forceinline__ v16bf frag_a(const bf16* row, int kbase) {
  FragU u;
  u.q[0] = *(const uint4*)(row + kbase);
  u.q[1] = *(const uint4*)(row + kbase + 8);
  return u.v;
}
// B fragment (32x16 bf16): lane col = lane&15; elems 0..15 = k..k+15 contiguous
__device__ __forceinline__ v16bf frag_b(const bf16* row, int kbase) {
  FragU u;
  u.q[0] = *(const uint4*)(row + kbase);
  u.q[1] = *(const uint4*)(row + kbase + 8);
  return u.v;
}

#define WMMA_BF16(a, b, c) \
  __builtin_amdgcn_wmma_f32_16x16x32_bf16(false, (a), false, (b), (short)0, (c), false, false)

// ---------------- Stage 1: h1 = W1*x + b1 (bf16 store) + GN partial stats --------------
__global__ __launch_bounds__(256) void k_proj1(const float* __restrict__ x,
                                               const float* __restrict__ W1,
                                               const float* __restrict__ b1,
                                               bf16* __restrict__ h1,
                                               float* __restrict__ gsum,
                                               float* __restrict__ gsumsq) {
  __shared__ __align__(16) bf16 Wl[CCH * LSTR];   // Wl[o][k]
  __shared__ __align__(16) bf16 Bl[CCH * LSTR];   // Bl[p][k]  (token-major)
  const int tid = threadIdx.x;
  const int n   = blockIdx.y;
  const int p0  = blockIdx.x * 128;

  for (int e = 0; e < 64; ++e) {                  // W1 (128x128) -> LDS bf16
    int lin = tid + 256 * e;
    int o = lin >> 7, c = lin & 127;
    Wl[o * LSTR + c] = (bf16)W1[o * CCH + c];
  }
  const float* xp = x + (size_t)n * CCH * HWPX + p0;
  for (int e = 0; e < 64; ++e) {                  // x tile (128c x 128p), transpose into [p][c]
    int lin = tid + 256 * e;
    int c = lin >> 7, p = lin & 127;
    Bl[p * LSTR + c] = (bf16)xp[(size_t)c * HWPX + p];
  }
  __syncthreads();

  const int wave = tid >> 5, lane = tid & 31;
  const int ob   = wave * 16;                     // o-tile == GroupNorm group `wave`
  const int colp = lane & 15;
  const int hi   = (lane & 16) ? 1 : 0;

  v8f acc[8] = {};
  for (int k0 = 0; k0 < CCH; k0 += 32) {
    v16bf a = frag_a(&Wl[(ob + colp) * LSTR], k0 + hi * 16);
    for (int j = 0; j < 8; ++j) {
      v16bf b = frag_b(&Bl[(j * 16 + colp) * LSTR], k0 + hi * 16);
      acc[j] = WMMA_BF16(a, b, acc[j]);
    }
  }

  float s = 0.f, s2 = 0.f;
  bf16* hp = h1 + (size_t)n * CCH * HWPX + p0;
  for (int r = 0; r < 8; ++r) {
    int o = ob + r + hi * 8;
    float bias = b1[o];
    for (int j = 0; j < 8; ++j) {
      float v = acc[j][r] + bias;
      s += v; s2 += v * v;
      hp[(size_t)o * HWPX + j * 16 + colp] = (bf16)v;
    }
  }
  for (int off = 16; off; off >>= 1) {
    s  += __shfl_down(s,  off, 32);
    s2 += __shfl_down(s2, off, 32);
  }
  if (lane == 0) {
    atomicAdd(&gsum[n * NGRP + wave], s);
    atomicAdd(&gsumsq[n * NGRP + wave], s2);
  }
}

// ---------------- Stage 2: fold GN stats with gn_w/gn_b into per-(n,c) affine ----------
__global__ void k_stats(const float* __restrict__ gsum, const float* __restrict__ gsumsq,
                        const float* __restrict__ gw, const float* __restrict__ gb,
                        float* __restrict__ scale, float* __restrict__ shift) {
  int i = blockIdx.x * blockDim.x + threadIdx.x;
  if (i >= NIMG * CCH) return;
  int n = i >> 7, c = i & 127, g = c >> 4;
  const float cnt = (float)CPG * (float)HWPX;
  float mu   = gsum[n * NGRP + g] / cnt;
  float var  = gsumsq[n * NGRP + g] / cnt - mu * mu;
  float rstd = rsqrtf(var + 1e-5f);
  float sc   = gw[c] * rstd;
  scale[i] = sc;
  shift[i] = gb[c] - mu * sc;
}

// ---------------- Stage 3: feats = W2*gelu(gn(h1)) + b2 (fp32 out) ---------------------
__global__ __launch_bounds__(256) void k_proj2(const bf16* __restrict__ h1,
                                               const float* __restrict__ W2,
                                               const float* __restrict__ b2,
                                               const float* __restrict__ scale,
                                               const float* __restrict__ shift,
                                               float* __restrict__ out) {
  __shared__ __align__(16) bf16 Wl[CCH * LSTR];
  __shared__ __align__(16) bf16 Bl[CCH * LSTR];
  const int tid = threadIdx.x;
  const int n   = blockIdx.y;
  const int p0  = blockIdx.x * 128;

  for (int e = 0; e < 64; ++e) {
    int lin = tid + 256 * e;
    int o = lin >> 7, c = lin & 127;
    Wl[o * LSTR + c] = (bf16)W2[o * CCH + c];
  }
  const bf16*  hp = h1 + (size_t)n * CCH * HWPX + p0;
  const float* scn = scale + n * CCH;
  const float* shn = shift + n * CCH;
  for (int e = 0; e < 64; ++e) {
    int lin = tid + 256 * e;
    int c = lin >> 7, p = lin & 127;
    float v = (float)hp[(size_t)c * HWPX + p];
    float t = v * scn[c] + shn[c];
    float g = 0.5f * t * (1.0f + erff(t * 0.70710678118654752f));  // exact GELU
    Bl[p * LSTR + c] = (bf16)g;
  }
  __syncthreads();

  const int wave = tid >> 5, lane = tid & 31;
  const int ob   = wave * 16;
  const int colp = lane & 15;
  const int hi   = (lane & 16) ? 1 : 0;

  v8f acc[8] = {};
  for (int k0 = 0; k0 < CCH; k0 += 32) {
    v16bf a = frag_a(&Wl[(ob + colp) * LSTR], k0 + hi * 16);
    for (int j = 0; j < 8; ++j) {
      v16bf b = frag_b(&Bl[(j * 16 + colp) * LSTR], k0 + hi * 16);
      acc[j] = WMMA_BF16(a, b, acc[j]);
    }
  }
  float* op = out + (size_t)n * CCH * HWPX + p0;
  for (int r = 0; r < 8; ++r) {
    int o = ob + r + hi * 8;
    float bias = b2[o];
    for (int j = 0; j < 8; ++j)
      op[(size_t)o * HWPX + j * 16 + colp] = acc[j][r] + bias;
  }
}

// ---------------- Stage 4: per-window Gram + log-softmax diag --------------------------
__global__ __launch_bounds__(128) void k_loss(const float* __restrict__ feats,
                                              const float* __restrict__ temp,
                                              float* __restrict__ lossAcc) {
  __shared__ __align__(16) bf16 Fl[64 * LSTR];   // Fl[token][c]
  __shared__ float Sl[64 * 65];                  // sim with +1 pad
  __shared__ float red[64];
  const int tid = threadIdx.x;
  const int wid = blockIdx.x;
  const int n = wid >> 10, rem = wid & 1023;
  const int wy = rem >> 5, wx = rem & 31;

  const float* fp = feats + (size_t)n * CCH * HWPX + (size_t)(wy * 8) * WPIX + wx * 8;
  for (int e = 0; e < 64; ++e) {                 // 64 tokens x 128 c
    int lin = tid + 128 * e;
    int c = lin >> 6, t = lin & 63;
    int ty = t >> 3, tx = t & 7;
    Fl[t * LSTR + c] = (bf16)fp[(size_t)c * HWPX + ty * WPIX + tx];
  }
  __syncthreads();

  const int wave = tid >> 5, lane = tid & 31;    // 4 waves x 16 sim-rows
  const int rb   = wave * 16;
  const int colp = lane & 15;
  const int hi   = (lane & 16) ? 1 : 0;

  v8f acc[4] = {};
  for (int k0 = 0; k0 < CCH; k0 += 32) {
    v16bf a = frag_a(&Fl[(rb + colp) * LSTR], k0 + hi * 16);
    for (int j = 0; j < 4; ++j) {
      v16bf b = frag_b(&Fl[(j * 16 + colp) * LSTR], k0 + hi * 16);
      acc[j] = WMMA_BF16(a, b, acc[j]);
    }
  }
  const float invT = 1.0f / temp[0];
  for (int j = 0; j < 4; ++j)
    for (int r = 0; r < 8; ++r)
      Sl[(rb + r + hi * 8) * 65 + j * 16 + colp] = acc[j][r] * invT;
  __syncthreads();

  if (tid < 64) {
    float m = -1e30f;
    for (int q = 0; q < 64; ++q) m = fmaxf(m, Sl[tid * 65 + q]);
    float se = 0.f;
    for (int q = 0; q < 64; ++q) se += expf(Sl[tid * 65 + q] - m);
    float lse = logf(se) + m;
    red[tid] = Sl[tid * 65 + tid] - lse;         // diagonal log-softmax
  }
  __syncthreads();
  if (tid == 0) {
    float a = 0.f;
    for (int q = 0; q < 64; ++q) a += red[q];
    atomicAdd(lossAcc, a);
  }
}

__global__ void k_final(const float* __restrict__ lossAcc, float* __restrict__ outLoss) {
  *outLoss = -(*lossAcc) / (10240.0f * 64.0f);
}

extern "C" void kernel_launch(void* const* d_in, const int* in_sizes, int n_in,
                              void* d_out, int out_size, void* d_ws, size_t ws_size,
                              hipStream_t stream) {
  const float* x    = (const float*)d_in[0];
  const float* W1   = (const float*)d_in[1];
  const float* b1   = (const float*)d_in[2];
  const float* gw   = (const float*)d_in[3];
  const float* gb   = (const float*)d_in[4];
  const float* W2   = (const float*)d_in[5];
  const float* b2   = (const float*)d_in[6];
  const float* temp = (const float*)d_in[7];
  float* out = (float*)d_out;

  float* wsf     = (float*)d_ws;
  float* lossAcc = wsf;            // 1 float
  float* gsum    = wsf + 16;       // 80 floats
  float* gsumsq  = wsf + 112;      // 80 floats
  float* scale   = wsf + 256;      // 1280 floats
  float* shiftv  = wsf + 1536;     // 1280 floats
  bf16*  h1      = (bf16*)((char*)d_ws + 16384);   // 10*128*65536 bf16 = 160 MB

  hipMemsetAsync(d_ws, 0, 16384, stream);          // zero loss accumulator + stats

  dim3 gproj(HWPX / 128, NIMG);                    // 512 pixel-tiles x 10 images
  k_proj1<<<gproj, 256, 0, stream>>>(x, W1, b1, h1, gsum, gsumsq);
  k_stats<<<NIMG, 128, 0, stream>>>(gsum, gsumsq, gw, gb, scale, shiftv);
  k_proj2<<<gproj, 256, 0, stream>>>(h1, W2, b2, scale, shiftv, out);
  k_loss<<<NIMG * 1024, 128, 0, stream>>>(out, temp, lossAcc);
  k_final<<<1, 1, 0, stream>>>(lossAcc, out + (size_t)NIMG * CCH * HWPX);
}
